// SecondOrderChannelAttension_15358803050800
// MI455X (gfx1250) — matrix operand order repaired and checked by reading, not compiled, not run on hardware
//
#include <hip/hip_runtime.h>

typedef __attribute__((ext_vector_type(16))) __bf16 v16bf;
typedef __attribute__((ext_vector_type(8)))  __bf16 v8bf;
typedef __attribute__((ext_vector_type(8)))  float  v8f;

#define BB 32
#define CC 64
#define HW 9216              // 96*96
#define MCHUNK 32
#define NCHUNK (HW / MCHUNK) // 288
#define SPLIT 16
#define CPS (NCHUNK / SPLIT) // 18 chunks per workgroup
#define LDB 40               // bf16-plane row stride (80 B, 16B-aligned slices)
#define LDM 66               // padded LDS row stride for 64x64 f32 matrices

union frag16 { v16bf v; v8bf h[2]; };

// ---- helpers -------------------------------------------------------------

__device__ __forceinline__ void split2(float f, __bf16 &h, __bf16 &l) {
    h = (__bf16)f;
    l = (__bf16)(f - (float)h);
}

// fp32-ish product via 3 bf16 WMMAs: hi*hi + hi*lo + lo*hi
__device__ __forceinline__ v8f wmma_bf16x2(v16bf ah, v16bf al, v16bf bh, v16bf bl, v8f acc) {
    acc = __builtin_amdgcn_wmma_f32_16x16x32_bf16(false, ah, false, bh, (short)0, acc, false, false);
    acc = __builtin_amdgcn_wmma_f32_16x16x32_bf16(false, ah, false, bl, (short)0, acc, false, false);
    acc = __builtin_amdgcn_wmma_f32_16x16x32_bf16(false, al, false, bh, (short)0, acc, false, false);
    return acc;
}

// 16x16 tile of D = op(A) * op(B), A,B 64x64 f32 in LDS (stride LDM), K=64.
// AMODE 1: a <- 0.5*(3I - a)   BMODE 1: b <- (3I - b)
template <int AMODE, int BMODE>
__device__ __forceinline__ v8f mm_tile(const float *MA, const float *MB, int ci, int cj, int lane) {
    v8f acc = {};
    const int g = lane >> 4, r = lane & 15, n = lane & 15;
    const int ra = ci * 16 + r;
    const int nb = cj * 16 + n;
    for (int kc = 0; kc < CC; kc += 32) {
        v16bf ah, al, bh, bl;
#pragma unroll
        for (int e = 0; e < 16; ++e) {
            int k = kc + e + 8 * g + (e < 8 ? 0 : 8); // 16-bit A-frag layout
            float f = MA[ra * LDM + k];
            if (AMODE == 1) f = 0.5f * (((ra == k) ? 3.0f : 0.0f) - f);
            __bf16 h, l; split2(f, h, l);
            ah[e] = h; al[e] = l;
        }
#pragma unroll
        for (int e = 0; e < 16; ++e) {
            int k = kc + e + 8 * g + (e < 8 ? 0 : 8); // 16-bit B-frag layout
            float f = MB[k * LDM + nb];
            if (BMODE == 1) f = ((k == nb) ? 3.0f : 0.0f) - f;
            __bf16 h, l; split2(f, h, l);
            bh[e] = h; bl[e] = l;
        }
        acc = wmma_bf16x2(ah, al, bh, bl, acc);
    }
    return acc;
}

__device__ __forceinline__ void store_tile(float *M, int ci, int cj, int lane, v8f acc, float scale) {
    const int g = lane >> 4, n = lane & 15;
#pragma unroll
    for (int j = 0; j < 8; ++j)
        M[(ci * 16 + j + 8 * g) * LDM + cj * 16 + n] = acc[j] * scale;
}

// ---- kernel 0: zero accumulators ----------------------------------------

__global__ void k_zero(float *p, int n) {
    int i = blockIdx.x * blockDim.x + threadIdx.x;
    if (i < n) p[i] = 0.0f;
}

// ---- kernel 1: split-K Gram (sum x x^T) + channel sums -------------------
// x chunk is pre-split into bf16 hi/lo planes at load time; fragments are
// then pure 16B ds_load_b128 slices (A and B: k=e+8g / e+8+8g are
// contiguous within each 8-element half).

__global__ __launch_bounds__(256) void k_gram(const float *__restrict__ x,
                                              float *__restrict__ G,
                                              float *__restrict__ sums) {
    __shared__ __align__(16) __bf16 xh[CC * LDB];
    __shared__ __align__(16) __bf16 xl[CC * LDB];
    __shared__ float red[256];
    const int tid  = threadIdx.x;
    const int lane = tid & 31, wave = tid >> 5;
    const int b = blockIdx.x / SPLIT, sp = blockIdx.x % SPLIT;
    const int ci  = wave >> 1;          // A row-block 0..3
    const int cj0 = (wave & 1) * 2;     // two B col-blocks per wave
    const int g = lane >> 4, r = lane & 15, n = lane & 15;
    const int lc = tid >> 2, off = (tid & 3) * 8; // loader: 8 floats/thread
    const float *xb = x + (size_t)b * CC * HW;

    v8f acc0 = {}; v8f acc1 = {};
    float psum = 0.0f;

    for (int ch = 0; ch < CPS; ++ch) {
        const int m0 = (sp * CPS + ch) * MCHUNK;
        const float4 *src = (const float4 *)(xb + (size_t)lc * HW + m0 + off);
        float4 v0 = src[0], v1 = src[1];
        float fv[8] = {v0.x, v0.y, v0.z, v0.w, v1.x, v1.y, v1.z, v1.w};
        v8bf hv, lv;
#pragma unroll
        for (int i = 0; i < 8; ++i) {
            float f = fv[i];
            psum += f;
            __bf16 h, l; split2(f, h, l);
            hv[i] = h; lv[i] = l;
        }
        *(v8bf *)&xh[lc * LDB + off] = hv;
        *(v8bf *)&xl[lc * LDB + off] = lv;
        __syncthreads();

        const int arow  = (ci * 16 + r) * LDB;
        const int brow0 = (cj0 * 16 + n) * LDB;
        const int brow1 = ((cj0 + 1) * 16 + n) * LDB;
        frag16 ah, al, b0h, b0l, b1h, b1l;
        ah.h[0]  = *(const v8bf *)&xh[arow + 8 * g];
        ah.h[1]  = *(const v8bf *)&xh[arow + 16 + 8 * g];
        al.h[0]  = *(const v8bf *)&xl[arow + 8 * g];
        al.h[1]  = *(const v8bf *)&xl[arow + 16 + 8 * g];
        b0h.h[0] = *(const v8bf *)&xh[brow0 + 8 * g];
        b0h.h[1] = *(const v8bf *)&xh[brow0 + 16 + 8 * g];
        b0l.h[0] = *(const v8bf *)&xl[brow0 + 8 * g];
        b0l.h[1] = *(const v8bf *)&xl[brow0 + 16 + 8 * g];
        b1h.h[0] = *(const v8bf *)&xh[brow1 + 8 * g];
        b1h.h[1] = *(const v8bf *)&xh[brow1 + 16 + 8 * g];
        b1l.h[0] = *(const v8bf *)&xl[brow1 + 8 * g];
        b1l.h[1] = *(const v8bf *)&xl[brow1 + 16 + 8 * g];

        acc0 = wmma_bf16x2(ah.v, al.v, b0h.v, b0l.v, acc0);
        acc1 = wmma_bf16x2(ah.v, al.v, b1h.v, b1l.v, acc1);
        __syncthreads();
    }

    float *Gb = G + (size_t)b * CC * CC;
#pragma unroll
    for (int j = 0; j < 8; ++j) {
        int m = ci * 16 + j + 8 * g;
        atomicAdd(&Gb[m * CC + cj0 * 16 + n],       acc0[j]);
        atomicAdd(&Gb[m * CC + (cj0 + 1) * 16 + n], acc1[j]);
    }
    red[tid] = psum;
    __syncthreads();
    if (tid < CC) {
        float s = red[tid * 4] + red[tid * 4 + 1] + red[tid * 4 + 2] + red[tid * 4 + 3];
        atomicAdd(&sums[b * CC + tid], s);
    }
}

// ---- kernel 2: cov finalize + Newton-Schulz sqrt + gating MLP ------------

__global__ __launch_bounds__(512) void k_ns(const float *__restrict__ G,
                                            const float *__restrict__ sums,
                                            const float *__restrict__ w1,
                                            const float *__restrict__ b1,
                                            const float *__restrict__ w2,
                                            const float *__restrict__ b2,
                                            float *__restrict__ gate) {
    __shared__ float MY[CC * LDM], MZ[CC * LDM], MW[CC * LDM];
    __shared__ float meanv[CC], svec[CC], hvec[8], rbuf[CC], scal[2];
    const int tid  = threadIdx.x;
    const int lane = tid & 31, wave = tid >> 5;
    const int ci = wave >> 2, cj = wave & 3;   // 16 waves -> 16 tiles
    const int b = blockIdx.x;

    if (tid < CC) {
        meanv[tid] = sums[b * CC + tid] * (1.0f / HW);
        svec[tid]  = 0.0f;
    }
    __syncthreads();

    const float *Gb = G + (size_t)b * CC * CC;
    // cov -> MZ
#pragma unroll
    for (int i = 0; i < 8; ++i) {
        int idx = tid * 8 + i;
        int c = idx >> 6, d = idx & 63;
        MZ[c * LDM + d] = Gb[idx] * (1.0f / HW) - meanv[c] * meanv[d];
    }
    __syncthreads();
    if (tid < CC) rbuf[tid] = MZ[tid * LDM + tid];
    __syncthreads();
    if (tid == 0) {
        float tr = 0.0f;
        for (int i = 0; i < CC; ++i) tr += rbuf[i];
        scal[0] = 1.0f / tr;
        scal[1] = sqrtf(tr);
    }
    __syncthreads();
    const float rnorm = scal[0];

    // MZ = A = cov/tr ; MW = ZY0 = 0.5*(3I - A)
#pragma unroll
    for (int i = 0; i < 8; ++i) {
        int idx = tid * 8 + i;
        int c = idx >> 6, d = idx & 63;
        float a = MZ[c * LDM + d] * rnorm;
        MZ[c * LDM + d] = a;
        MW[c * LDM + d] = 0.5f * (((c == d) ? 3.0f : 0.0f) - a);
    }
    __syncthreads();

    // Y = A @ ZY0
    { v8f t = mm_tile<0, 0>(MZ, MW, ci, cj, lane); store_tile(MY, ci, cj, lane, t, 1.0f); }
    __syncthreads();
    // Z = ZY0 (copy W -> Z)
#pragma unroll
    for (int i = 0; i < 8; ++i) {
        int idx = tid * 8 + i;
        int c = idx >> 6, d = idx & 63;
        MZ[c * LDM + d] = MW[c * LDM + d];
    }
    __syncthreads();

    for (int it = 0; it < 3; ++it) {
        // W = 0.5*(3I - Z) @ Y   (transform fused into A-frag build)
        v8f tw = mm_tile<1, 0>(MZ, MY, ci, cj, lane);
        store_tile(MW, ci, cj, lane, tw, 1.0f);
        __syncthreads();
        // Ynew = Y@W ; Znew = W@Z (both from old values, then commit)
        v8f ty = mm_tile<0, 0>(MY, MW, ci, cj, lane);
        v8f tz = mm_tile<0, 0>(MW, MZ, ci, cj, lane);
        __syncthreads();
        store_tile(MY, ci, cj, lane, ty, 1.0f);
        store_tile(MZ, ci, cj, lane, tz, 1.0f);
        __syncthreads();
    }

    // P = Z@Y -> MW
    {
        v8f tp = mm_tile<0, 0>(MZ, MY, ci, cj, lane);
        store_tile(MW, ci, cj, lane, tp, 1.0f);
        __syncthreads();
    }
    // cov_sqrt tile = 0.5*sqrt(tr) * Y @ (3I - P); reduce mean over rows (dim=1)
    const float fs = 0.5f * scal[1];
    v8f ts = mm_tile<0, 1>(MY, MW, ci, cj, lane);
    float part = 0.0f;
#pragma unroll
    for (int j = 0; j < 8; ++j) part += ts[j];
    part *= fs;
    atomicAdd(&svec[cj * 16 + (lane & 15)], part);
    __syncthreads();

    if (tid < 8) {
        float acc = b1[tid];
        for (int d = 0; d < CC; ++d) acc += svec[d] * (1.0f / CC) * w1[tid * CC + d];
        hvec[tid] = acc > 0.0f ? acc : 0.0f;
    }
    __syncthreads();
    if (tid < CC) {
        float acc = b2[tid];
        for (int k = 0; k < 8; ++k) acc += hvec[k] * w2[tid * 8 + k];
        gate[b * CC + tid] = 1.0f / (1.0f + __expf(-acc));
    }
}

// ---- kernel 3: out = x * gate[b][c] --------------------------------------

__global__ __launch_bounds__(256) void k_apply(const float *__restrict__ x,
                                               const float *__restrict__ gate,
                                               float *__restrict__ out) {
    const int bc = blockIdx.x;
    const float gv = gate[bc];
    const float4 *xi = (const float4 *)(x + (size_t)bc * HW);
    float4 *oo = (float4 *)(out + (size_t)bc * HW);
    for (int i = threadIdx.x; i < HW / 4; i += blockDim.x) {
        float4 v = xi[i];
        v.x *= gv; v.y *= gv; v.z *= gv; v.w *= gv;
        oo[i] = v;
    }
}

// ---- launcher ------------------------------------------------------------

extern "C" void kernel_launch(void *const *d_in, const int *in_sizes, int n_in,
                              void *d_out, int out_size, void *d_ws, size_t ws_size,
                              hipStream_t stream) {
    const float *x  = (const float *)d_in[0];
    const float *w1 = (const float *)d_in[1];
    const float *b1 = (const float *)d_in[2];
    const float *w2 = (const float *)d_in[3];
    const float *b2 = (const float *)d_in[4];

    float *G    = (float *)d_ws;        // 32*64*64
    float *sums = G + BB * CC * CC;     // 32*64
    float *gate = sums + BB * CC;       // 32*64

    const int nz = BB * CC * CC + BB * CC;
    k_zero<<<(nz + 255) / 256, 256, 0, stream>>>(G, nz);
    k_gram<<<BB * SPLIT, 256, 0, stream>>>(x, G, sums);
    k_ns<<<BB, 512, 0, stream>>>(G, sums, w1, b1, w2, b2, gate);
    k_apply<<<BB * CC, 256, 0, stream>>>(x, gate, (float *)d_out);
}
